// WASTGN_56788057588042
// MI455X (gfx1250) — matrix operand order repaired and checked by reading, not compiled, not run on hardware
//
#include <hip/hip_runtime.h>
#include <hip/hip_bf16.h>

// ---------------------------------------------------------------------------
// WASTGN forward for MI455X (gfx1250, wave32, WMMA f32_16x16x32_f16)
// All GEMM operands f16 (f32 accumulate); fragment I/O vectorized to b128;
// tile staging global->LDS via Tensor Data Mover where available.
// ---------------------------------------------------------------------------
#define B_   32
#define T_   168
#define N_   64
#define F_   18
#define TC_  64
#define SP_  64
#define ED_  8
#define NH_  24
#define NT_  6
#define OUTC_ (NH_*NT_)           // 144
#define KOUT_ (TC_*T_)            // 10752
#define ROWS_ ((size_t)B_*T_*N_)  // 344064

typedef __attribute__((ext_vector_type(16))) _Float16 v16h;
typedef __attribute__((ext_vector_type(8)))  float    v8f;
typedef unsigned int __attribute__((ext_vector_type(4))) v4u;
typedef int          __attribute__((ext_vector_type(8))) v8i;
typedef int          __attribute__((ext_vector_type(4))) v4i;

#if defined(__has_builtin)
#if __has_builtin(__builtin_amdgcn_tensor_load_to_lds) && __has_builtin(__builtin_amdgcn_s_wait_tensorcnt)
#define HAVE_TDM 1
#endif
#endif
#ifndef HAVE_TDM
#define HAVE_TDM 0
#endif

// ----- WMMA helpers (CDNA5 wave32 fragment layouts) ------------------------
__device__ __forceinline__ v8f wmma_f16(v16h a, v16h b, v8f c) {
  return __builtin_amdgcn_wmma_f32_16x16x32_f16(
      false, a, false, b, (short)0, c, false, false);
}

// A fragment (16x32 f16) from row-major storage, ld in halves.
__device__ __forceinline__ v16h load_a_frag(const _Float16* src, int ld) {
  int lane = threadIdx.x & 31;
  const _Float16* p = src + (lane & 15) * ld + ((lane >> 4) << 4);
  v16h a;
  ((int4*)&a)[0] = *(const int4*)(p);
  ((int4*)&a)[1] = *(const int4*)(p + 8);
  return a;
}

// B fragment (32x16 f16, K x N) from N-major storage: element (k,n) at src[n*ldn+k].
__device__ __forceinline__ v16h load_bT_frag(const _Float16* src, int ldn) {
  int lane = threadIdx.x & 31;
  const _Float16* p = src + (lane & 15) * ldn + ((lane >> 4) << 4);
  v16h b;
  ((int4*)&b)[0] = *(const int4*)(p);
  ((int4*)&b)[1] = *(const int4*)(p + 8);
  return b;
}

__device__ __forceinline__ float sigm(float v) { return 1.f / (1.f + __expf(-v)); }

#if HAVE_TDM
// Issue one TDM 2D tile load (f16 elements) global->LDS. Dims/strides in elements.
// D# layout per CDNA5 ISA ch.8: group0 = {count, lds_addr, global_addr, type=2},
// group1 = {data_size=1(2B), tensor_dim0/1, tile_dim0/1, dim0_stride}.
// This toolchain's builtin takes 6 args: (g0 u32x4, g1 i32x8, g2 i32x4, g3 i32x4,
// extra i32x8, cpol i32) -- groups 2/3 unused for 2D tensors, pass zeros.
__device__ __forceinline__ void tdm_load_tile_f16(const void* gptr, void* lptr,
                                                  int td0, int td1, int t0, int t1,
                                                  int stride0) {
  unsigned long long ga = (unsigned long long)(uintptr_t)gptr;
  unsigned lds = (unsigned)(uintptr_t)lptr;        // LDS byte offset in low 32 bits
  v4u g0;
  g0[0] = 1u;                                      // count=1, user descriptor
  g0[1] = lds;                                     // lds_addr (bytes)
  g0[2] = (unsigned)(ga & 0xFFFFFFFFu);            // global_addr[31:0]
  g0[3] = (unsigned)((ga >> 32) & 0x01FFFFFFu) | (2u << 30);  // addr[56:32] | type=2
  v8i g1;
  g1[0] = 1 << 16;                                 // workgroup_mask=0, data_size=1 (2B)
  g1[1] = (td0 & 0xFFFF) << 16;                    // tensor_dim0[15:0] -> bits63:48
  g1[2] = ((td0 >> 16) & 0xFFFF) | ((td1 & 0xFFFF) << 16);    // dim0 hi | dim1 lo
  g1[3] = ((td1 >> 16) & 0xFFFF) | ((t0 & 0xFFFF) << 16);     // dim1 hi | tile_dim0
  g1[4] = (t1 & 0xFFFF);                           // tile_dim1 (tile_dim2 = 0)
  g1[5] = stride0;                                 // tensor_dim0_stride[31:0]
  g1[6] = 0;
  g1[7] = 0;
  v4i z4 = {0, 0, 0, 0};
  v8i z8 = {0, 0, 0, 0, 0, 0, 0, 0};
  __builtin_amdgcn_tensor_load_to_lds(g0, g1, z4, z4, z8, 0);
}
#endif

// ---------------------------------------------------------------------------
// Setup kernels: pack weights into f16 N-major fragment-friendly layouts
// ---------------------------------------------------------------------------
__global__ void k_pack_conv_w(const float* __restrict__ wf, const float* __restrict__ wg,
                              _Float16* __restrict__ dst, int cin, int KP) {
  int total = 2 * 3 * 64 * KP;
  for (int i = blockIdx.x * 256 + threadIdx.x; i < total; i += gridDim.x * 256) {
    int ci  = i % KP;
    int c   = (i / KP) % 64;
    int tap = (i / (KP * 64)) % 3;
    int mat = i / (KP * 64 * 3);
    const float* w = mat ? wg : wf;
    dst[i] = (ci < cin) ? (_Float16)w[(c * cin + ci) * 3 + tap] : (_Float16)0.f;
  }
}

__global__ void k_pack_spw(const float* __restrict__ spW, _Float16* __restrict__ dst) {
  int i = blockIdx.x * 256 + threadIdx.x;
  if (i < TC_ * SP_) { int s = i >> 6, c = i & 63; dst[i] = (_Float16)spW[c * SP_ + s]; }
}

__global__ void k_w_to_f16(const float* __restrict__ ow, _Float16* __restrict__ dst) {
  size_t i = (size_t)blockIdx.x * 256 + threadIdx.x;
  if (i < (size_t)OUTC_ * KOUT_) dst[i] = (_Float16)ow[i];
}

// ---------------------------------------------------------------------------
// K1: static base adjacency (64x64 f32)
// ---------------------------------------------------------------------------
__global__ void k_static_base(const float* __restrict__ E1, const float* __restrict__ E2,
                              const float* __restrict__ fixed_adj,
                              const float* __restrict__ L_norm,
                              const float* __restrict__ gamma_p,
                              const float* __restrict__ kappa,
                              float* __restrict__ A_base) {
  int n = threadIdx.x;             // one block, 64 threads
  float km = 0.f;
  for (int f = 0; f < F_; ++f) km += sigm(kappa[f]);
  km /= (float)F_;
  float sg = sigm(gamma_p[0]);
  float e1[ED_];
#pragma unroll
  for (int d = 0; d < ED_; ++d) e1[d] = E1[n * ED_ + d];
  float logit[N_];
  float mx = -1e30f;
  for (int m = 0; m < N_; ++m) {
    float s = 0.f;
#pragma unroll
    for (int d = 0; d < ED_; ++d) s += e1[d] * E2[m * ED_ + d];
    s = fmaxf(s, 0.f);
    logit[m] = s;
    mx = fmaxf(mx, s);
  }
  float den = 0.f;
  for (int m = 0; m < N_; ++m) { logit[m] = __expf(logit[m] - mx); den += logit[m]; }
  for (int m = 0; m < N_; ++m) {
    float adp = logit[m] / den;
    float idn = (m == n) ? 1.f : 0.f;
    A_base[n * N_ + m] = fixed_adj[n * N_ + m] + adp + sg * km * (idn - L_norm[n * N_ + m]);
  }
}

// ---------------------------------------------------------------------------
// K2: dynamic wind adjacency per (b,t), f16 output (b128 stores)
// ---------------------------------------------------------------------------
__global__ void k_dyn_adj(const float* __restrict__ x,
                          const float* __restrict__ dir_lon, const float* __restrict__ dir_lat,
                          const float* __restrict__ inv_dist, const float* __restrict__ A_base,
                          const float* __restrict__ alpha_p, const float* __restrict__ beta_p,
                          _Float16* __restrict__ adjA) {
  int bt = blockIdx.x;
  __shared__ float swu[N_], swv[N_], spr[N_];
  __shared__ float srow[N_][4];
  int tid = threadIdx.x;           // 256
  if (tid < N_) {
    const float* xr = x + ((size_t)bt * N_ + tid) * F_;
    swu[tid] = xr[12]; swv[tid] = xr[13]; spr[tid] = xr[10];
  }
  __syncthreads();
  float sa = sigm(alpha_p[0]);
  float sb = sigm(beta_p[0]);
  int row  = tid >> 2;
  int cseg = tid & 3;
  float e[16];
  float rs = 0.f;
#pragma unroll
  for (int j = 0; j < 16; ++j) {
    int col = cseg * 16 + j;
    float dl = dir_lon[row * N_ + col], dt = dir_lat[row * N_ + col];
    float push = swu[row] * dl + swv[row] * dt;
    float recv = -(swu[col] * dl + swv[col] * dt);
    float adv  = fmaxf(push + recv, 0.f) * 0.5f;
    float pm   = sigm(sb * (spr[row] - spr[col]));
    float v    = adv * pm * inv_dist[row * N_ + col];
    e[j] = v; rs += v;
  }
  srow[row][cseg] = rs;
  __syncthreads();
  float rsum = srow[row][0] + srow[row][1] + srow[row][2] + srow[row][3] + 1e-8f;
  _Float16 t16[16];
#pragma unroll
  for (int j = 0; j < 16; ++j) {
    int col = cseg * 16 + j;
    t16[j] = (_Float16)(A_base[row * N_ + col] + sa * e[j] / rsum);
  }
  _Float16* out = adjA + ((size_t)bt * N_ + row) * N_ + cseg * 16;
  ((int4*)out)[0] = ((const int4*)t16)[0];
  ((int4*)out)[1] = ((const int4*)t16)[1];
}

// ---------------------------------------------------------------------------
// K3: x f32 (rows of 18) -> f16 padded to 32 channels; one int4 store/thread
// ---------------------------------------------------------------------------
__global__ void k_x_to_f16(const float* __restrict__ x, _Float16* __restrict__ x16) {
  size_t i = (size_t)blockIdx.x * 256 + threadIdx.x;
  if (i >= ROWS_ * 4) return;
  int ch = (int)(i & 3);
  size_t r = i >> 2;
  const float* xr = x + r * F_;
  _Float16 t8[8];
#pragma unroll
  for (int j = 0; j < 8; ++j) {
    int k = ch * 8 + j;
    t8[j] = (k < F_) ? (_Float16)xr[k] : (_Float16)0.f;
  }
  ((int4*)x16)[i] = *(const int4*)t8;
}

// ---------------------------------------------------------------------------
// K4: temporal GLU conv = 3 shifted WMMA GEMMs; Y=(X*Wf+bf)*sigmoid(X*Wg+bg)
//     Packed weights staged to LDS via TDM (fallback: b128 copies).
// ---------------------------------------------------------------------------
template <int KP, bool OUT16>
__global__ void k_tconv_glu(const _Float16* __restrict__ Xin,
                            const _Float16* __restrict__ Wpk,
                            const float* __restrict__ bf, const float* __restrict__ bg,
                            void* __restrict__ Yout) {
  __shared__ _Float16 sW[2 * 3 * 64 * KP];
  int tid = threadIdx.x;
#if HAVE_TDM
  if (tid < 32) {
    tdm_load_tile_f16(Wpk, sW, 2 * 3 * 64 * KP, 1, 2 * 3 * 64 * KP, 1, 2 * 3 * 64 * KP);
    __builtin_amdgcn_s_wait_tensorcnt(0);
  }
  __syncthreads();
#else
  const int n4 = (2 * 3 * 64 * KP) / 8;
  for (int i = tid; i < n4; i += 256) ((int4*)sW)[i] = ((const int4*)Wpk)[i];
  __syncthreads();
#endif

  int wave  = tid >> 5;
  int mtile = blockIdx.x * 8 + wave;
  size_t rowbase = (size_t)mtile * 16;
  int t = (int)((rowbase / N_) % T_);   // uniform within tile (16 | N)

  v8f accF[4], accG[4];
#pragma unroll
  for (int nt = 0; nt < 4; ++nt) {
    accF[nt] = (v8f){0.f,0.f,0.f,0.f,0.f,0.f,0.f,0.f};
    accG[nt] = (v8f){0.f,0.f,0.f,0.f,0.f,0.f,0.f,0.f};
  }
#pragma unroll
  for (int tap = 0; tap < 3; ++tap) {
    int tt = t + tap - 1;
    if (tt < 0 || tt >= T_) continue;                  // wave-uniform zero-pad
    const _Float16* Arow = Xin + (rowbase + (size_t)(tap - 1) * N_) * KP;
#pragma unroll
    for (int ks = 0; ks < KP / 32; ++ks) {
      v16h a = load_a_frag(Arow + ks * 32, KP);
#pragma unroll
      for (int nt = 0; nt < 4; ++nt) {
        const _Float16* bfp = sW + ((0 * 3 + tap) * 64 + nt * 16) * KP + ks * 32;
        const _Float16* bgp = sW + ((1 * 3 + tap) * 64 + nt * 16) * KP + ks * 32;
        accF[nt] = wmma_f16(a, load_bT_frag(bfp, KP), accF[nt]);
        accG[nt] = wmma_f16(a, load_bT_frag(bgp, KP), accG[nt]);
      }
    }
  }
  int lane = tid & 31;
  int nloc = lane & 15;
  int mo   = (lane >> 4) << 3;
#pragma unroll
  for (int nt = 0; nt < 4; ++nt) {
    int c = nt * 16 + nloc;
    float biasF = bf[c], biasG = bg[c];
#pragma unroll
    for (int i = 0; i < 8; ++i) {
      float f = accF[nt][i] + biasF;
      float g = accG[nt][i] + biasG;
      float y = f * sigm(g);
      size_t row = rowbase + mo + i;
      if (OUT16) ((_Float16*)Yout)[row * TC_ + c] = (_Float16)y;
      else       ((float*)Yout)[row * TC_ + c]    = y;
    }
  }
}

// ---------------------------------------------------------------------------
// K5: fused graph conv + spatial linear per (b,t), reassociated:
//     Q = relu( adj @ (P @ spW) + spb )
//     adj & P tiles staged via TDM (fallback b128); GEMM1 D-fragment stored
//     b128 into sIT[s][node] == GEMM2 B orientation.
// ---------------------------------------------------------------------------
__global__ void k_graph_spatial(const _Float16* __restrict__ adjA,
                                const _Float16* __restrict__ P,
                                const _Float16* __restrict__ spWt,
                                const float* __restrict__ spb,
                                _Float16* __restrict__ Q) {
  __shared__ _Float16 sAdj[N_ * N_];
  __shared__ _Float16 sP[N_ * TC_];
  __shared__ _Float16 sIT[SP_ * N_];   // [s][node]
  __shared__ _Float16 sW[SP_ * TC_];   // [s][c]
  int bt = blockIdx.x;
  int tid = threadIdx.x;               // 128
#if HAVE_TDM
  if (tid < 32) {
    tdm_load_tile_f16(adjA + (size_t)bt * N_ * N_, sAdj, N_, N_, N_, N_, N_);
    tdm_load_tile_f16(P + (size_t)bt * N_ * TC_, sP, TC_, N_, TC_, N_, TC_);
  }
  for (int i = tid; i < (SP_ * TC_) / 8; i += 128) ((int4*)sW)[i] = ((const int4*)spWt)[i];
  if (tid < 32) __builtin_amdgcn_s_wait_tensorcnt(0);
  __syncthreads();
#else
  for (int i = tid; i < (SP_ * TC_) / 8; i += 128) ((int4*)sW)[i] = ((const int4*)spWt)[i];
  const int4* ga = (const int4*)(adjA + (size_t)bt * N_ * N_);
  const int4* gp = (const int4*)(P + (size_t)bt * N_ * TC_);
  for (int i = tid; i < 512; i += 128) { ((int4*)sAdj)[i] = ga[i]; ((int4*)sP)[i] = gp[i]; }
  __syncthreads();
#endif

  int wave = tid >> 5, lane = tid & 31;
  int nloc = lane & 15, mo = (lane >> 4) << 3;

  // GEMM1: inter = P @ spW
  v8f acc[4];
#pragma unroll
  for (int nt = 0; nt < 4; ++nt) acc[nt] = (v8f){0.f,0.f,0.f,0.f,0.f,0.f,0.f,0.f};
#pragma unroll
  for (int ks = 0; ks < 2; ++ks) {
    v16h a = load_a_frag(sP + (wave * 16) * TC_ + ks * 32, TC_);
#pragma unroll
    for (int nt = 0; nt < 4; ++nt)
      acc[nt] = wmma_f16(a, load_bT_frag(sW + (nt * 16) * TC_ + ks * 32, TC_), acc[nt]);
  }
#pragma unroll
  for (int nt = 0; nt < 4; ++nt) {
    _Float16 t8[8];
#pragma unroll
    for (int i = 0; i < 8; ++i) t8[i] = (_Float16)acc[nt][i];
    *(int4*)(sIT + (nt * 16 + nloc) * N_ + wave * 16 + mo) = *(const int4*)t8;
  }
  __syncthreads();

  // GEMM2: out = adj @ inter
  v8f acc2[4];
#pragma unroll
  for (int nt = 0; nt < 4; ++nt) acc2[nt] = (v8f){0.f,0.f,0.f,0.f,0.f,0.f,0.f,0.f};
#pragma unroll
  for (int ks = 0; ks < 2; ++ks) {
    v16h a = load_a_frag(sAdj + (wave * 16) * N_ + ks * 32, N_);
#pragma unroll
    for (int nt = 0; nt < 4; ++nt)
      acc2[nt] = wmma_f16(a, load_bT_frag(sIT + (nt * 16) * N_ + ks * 32, N_), acc2[nt]);
  }
  _Float16* out = Q + (size_t)bt * N_ * SP_;
#pragma unroll
  for (int nt = 0; nt < 4; ++nt) {
    float bias = spb[nt * 16 + nloc];
#pragma unroll
    for (int i = 0; i < 8; ++i) {
      float v = fmaxf(acc2[nt][i] + bias, 0.f);
      out[(wave * 16 + mo + i) * SP_ + nt * 16 + nloc] = (_Float16)v;
    }
  }
}

// ---------------------------------------------------------------------------
// K6: LayerNorm(h + residual). One wave per row (64 ch, 2 per lane).
// ---------------------------------------------------------------------------
__global__ void k_ln(const float* __restrict__ S, const float* __restrict__ x,
                     const float* __restrict__ res_w, const float* __restrict__ res_b,
                     const float* __restrict__ h1f,
                     const float* __restrict__ lng, const float* __restrict__ lnb,
                     float* __restrict__ outF32, _Float16* __restrict__ outF16, int mode) {
  int wave = threadIdx.x >> 5;
  int lane = threadIdx.x & 31;
  size_t row = (size_t)blockIdx.x * 8 + wave;
  const float* srow = S + row * TC_;
  float v0 = srow[lane], v1 = srow[lane + 32];
  if (mode == 0) {
    const float* xr = x + row * F_;
    float r0 = res_b[lane], r1 = res_b[lane + 32];
#pragma unroll
    for (int f = 0; f < F_; ++f) {
      float xv = xr[f];
      r0 += xv * res_w[lane * F_ + f];
      r1 += xv * res_w[(lane + 32) * F_ + f];
    }
    v0 += r0; v1 += r1;
  } else {
    v0 += h1f[row * TC_ + lane];
    v1 += h1f[row * TC_ + lane + 32];
  }
  float s = v0 + v1;
#pragma unroll
  for (int o = 16; o > 0; o >>= 1) s += __shfl_xor(s, o, 32);
  float mu = s * (1.f / 64.f);
  float d0 = v0 - mu, d1 = v1 - mu;
  float q = d0 * d0 + d1 * d1;
#pragma unroll
  for (int o = 16; o > 0; o >>= 1) q += __shfl_xor(q, o, 32);
  float inv = rsqrtf(q * (1.f / 64.f) + 1e-5f);
  float y0 = d0 * inv * lng[lane] + lnb[lane];
  float y1 = d1 * inv * lng[lane + 32] + lnb[lane + 32];
  if (mode == 0) { outF32[row * TC_ + lane] = y0; outF32[row * TC_ + lane + 32] = y1; }
  outF16[row * TC_ + lane] = (_Float16)y0;
  outF16[row * TC_ + lane + 32] = (_Float16)y1;
}

// ---------------------------------------------------------------------------
// K7: final projection GEMM: M=B*N=2048, K=10752, N=144.
// ---------------------------------------------------------------------------
__global__ void k_out_gemm(const _Float16* __restrict__ H2,
                           const _Float16* __restrict__ ow16, const float* __restrict__ ob,
                           float* __restrict__ out) {
  __shared__ _Float16 sA[16 * 32];
  int mtile = blockIdx.x;
  int tid = threadIdx.x;
  int wave = tid >> 5, lane = tid & 31;
  int b  = (mtile * 16) >> 6;
  int n0 = (mtile * 16) & 63;
  int o  = wave * 16 + (lane & 15);
  v8f acc = (v8f){0.f,0.f,0.f,0.f,0.f,0.f,0.f,0.f};
  const _Float16* bp = ow16 + (size_t)o * KOUT_ + ((lane >> 4) << 4);
  for (int kk = 0; kk < KOUT_ / 32; ++kk) {
    __syncthreads();
    if (tid < 64) {
      int m = tid >> 2, j4 = tid & 3;
      int t = kk >> 1, c0 = (kk & 1) * 32;
      const int4* src = (const int4*)(H2 + (((size_t)b * T_ + t) * N_ + n0 + m) * TC_ + c0);
      ((int4*)sA)[tid] = src[j4];
    }
    __syncthreads();
    v16h a = load_a_frag(sA, 32);
    const _Float16* p = bp + kk * 32;
    v16h bfr;
    ((int4*)&bfr)[0] = *(const int4*)(p);
    ((int4*)&bfr)[1] = *(const int4*)(p + 8);
    __builtin_prefetch(p + 32, 0, 1);
    acc = wmma_f16(a, bfr, acc);
  }
  float bias = ob[o];
  int h = o / NT_, tp = o % NT_;
  int mo = (lane >> 4) << 3;
#pragma unroll
  for (int i = 0; i < 8; ++i) {
    int n = n0 + mo + i;
    out[(((size_t)b * NH_ + h) * N_ + n) * NT_ + tp] = acc[i] + bias;
  }
}

// ---------------------------------------------------------------------------
// Host launcher
// ---------------------------------------------------------------------------
extern "C" void kernel_launch(void* const* d_in, const int* in_sizes, int n_in,
                              void* d_out, int out_size, void* d_ws, size_t ws_size,
                              hipStream_t stream) {
  (void)in_sizes; (void)n_in; (void)out_size; (void)ws_size;
  const float* x        = (const float*)d_in[0];
  const float* E1       = (const float*)d_in[1];
  const float* E2       = (const float*)d_in[2];
  const float* alpha_p  = (const float*)d_in[3];
  const float* beta_p   = (const float*)d_in[4];
  const float* gamma_p  = (const float*)d_in[5];
  const float* kappa    = (const float*)d_in[6];
  const float* b1_t1f_w = (const float*)d_in[7];
  const float* b1_t1f_b = (const float*)d_in[8];
  const float* b1_t1g_w = (const float*)d_in[9];
  const float* b1_t1g_b = (const float*)d_in[10];
  const float* b1_sp_W  = (const float*)d_in[11];
  const float* b1_sp_b  = (const float*)d_in[12];
  const float* b1_t2f_w = (const float*)d_in[13];
  const float* b1_t2f_b = (const float*)d_in[14];
  const float* b1_t2g_w = (const float*)d_in[15];
  const float* b1_t2g_b = (const float*)d_in[16];
  const float* b1_ln_g  = (const float*)d_in[17];
  const float* b1_ln_b  = (const float*)d_in[18];
  const float* b1_res_w = (const float*)d_in[19];
  const float* b1_res_b = (const float*)d_in[20];
  const float* b2_t1f_w = (const float*)d_in[21];
  const float* b2_t1f_b = (const float*)d_in[22];
  const float* b2_t1g_w = (const float*)d_in[23];
  const float* b2_t1g_b = (const float*)d_in[24];
  const float* b2_sp_W  = (const float*)d_in[25];
  const float* b2_sp_b  = (const float*)d_in[26];
  const float* b2_t2f_w = (const float*)d_in[27];
  const float* b2_t2f_b = (const float*)d_in[28];
  const float* b2_t2g_w = (const float*)d_in[29];
  const float* b2_t2g_b = (const float*)d_in[30];
  const float* b2_ln_g  = (const float*)d_in[31];
  const float* b2_ln_b  = (const float*)d_in[32];
  const float* out_w    = (const float*)d_in[33];
  const float* out_b    = (const float*)d_in[34];
  const float* fixed_adj= (const float*)d_in[35];
  const float* dir_lon  = (const float*)d_in[36];
  const float* dir_lat  = (const float*)d_in[37];
  const float* inv_dist = (const float*)d_in[38];
  const float* L_norm   = (const float*)d_in[39];

  char* ws = (char*)d_ws;
  const size_t szAdj = ROWS_ * N_ * sizeof(_Float16);
  const size_t szX16 = ROWS_ * 32 * sizeof(_Float16);
  const size_t szH16 = ROWS_ * TC_ * sizeof(_Float16);
  const size_t szH32 = ROWS_ * TC_ * sizeof(float);
  size_t off = 0;
  _Float16* adjA  = (_Float16*)(ws + off); off += szAdj;
  _Float16* x16   = (_Float16*)(ws + off); off += szX16;
  _Float16* bufP  = (_Float16*)(ws + off); off += szH16;
  _Float16* bufQ  = (_Float16*)(ws + off); off += szH16;
  float*    bufS  = (float*)   (ws + off); off += szH32;
  float*    h1f32 = (float*)   (ws + off); off += szH32;
  _Float16* h1f16 = (_Float16*)(ws + off); off += szH16;
  _Float16* h2f16 = (_Float16*)(ws + off); off += szH16;
  float*    Abase = (float*)   (ws + off); off += 64 * 64 * sizeof(float);
  _Float16* wpk11 = (_Float16*)(ws + off); off += (size_t)2*3*64*32 * 2;
  _Float16* wpk12 = (_Float16*)(ws + off); off += (size_t)2*3*64*64 * 2;
  _Float16* wpk21 = (_Float16*)(ws + off); off += (size_t)2*3*64*64 * 2;
  _Float16* wpk22 = (_Float16*)(ws + off); off += (size_t)2*3*64*64 * 2;
  _Float16* spwt1 = (_Float16*)(ws + off); off += (size_t)TC_*SP_ * 2;
  _Float16* spwt2 = (_Float16*)(ws + off); off += (size_t)TC_*SP_ * 2;
  _Float16* ow16  = (_Float16*)(ws + off); off += (size_t)OUTC_*KOUT_ * 2;

  const int nBT     = B_ * T_;
  const int nTilesM = (int)(ROWS_ / 16) / 8;
  const int nLN     = (int)(ROWS_ / 8);
  const int nPad    = (int)(ROWS_ * 4 / 256);

  k_pack_conv_w<<<48, 256, 0, stream>>>(b1_t1f_w, b1_t1g_w, wpk11, F_, 32);
  k_pack_conv_w<<<96, 256, 0, stream>>>(b1_t2f_w, b1_t2g_w, wpk12, SP_, 64);
  k_pack_conv_w<<<96, 256, 0, stream>>>(b2_t1f_w, b2_t1g_w, wpk21, TC_, 64);
  k_pack_conv_w<<<96, 256, 0, stream>>>(b2_t2f_w, b2_t2g_w, wpk22, SP_, 64);
  k_pack_spw<<<16, 256, 0, stream>>>(b1_sp_W, spwt1);
  k_pack_spw<<<16, 256, 0, stream>>>(b2_sp_W, spwt2);
  k_w_to_f16<<<(int)(((size_t)OUTC_*KOUT_ + 255) / 256), 256, 0, stream>>>(out_w, ow16);

  k_static_base<<<1, 64, 0, stream>>>(E1, E2, fixed_adj, L_norm, gamma_p, kappa, Abase);
  k_dyn_adj<<<nBT, 256, 0, stream>>>(x, dir_lon, dir_lat, inv_dist, Abase, alpha_p, beta_p, adjA);
  k_x_to_f16<<<nPad, 256, 0, stream>>>(x, x16);

  k_tconv_glu<32, true ><<<nTilesM, 256, 0, stream>>>(x16,   wpk11, b1_t1f_b, b1_t1g_b, bufP);
  k_graph_spatial<<<nBT, 128, 0, stream>>>(adjA, bufP, spwt1, b1_sp_b, bufQ);
  k_tconv_glu<64, false><<<nTilesM, 256, 0, stream>>>(bufQ,  wpk12, b1_t2f_b, b1_t2g_b, bufS);
  k_ln<<<nLN, 256, 0, stream>>>(bufS, x, b1_res_w, b1_res_b, nullptr, b1_ln_g, b1_ln_b, h1f32, h1f16, 0);

  k_tconv_glu<64, true ><<<nTilesM, 256, 0, stream>>>(h1f16, wpk21, b2_t1f_b, b2_t1g_b, bufP);
  k_graph_spatial<<<nBT, 128, 0, stream>>>(adjA, bufP, spwt2, b2_sp_b, bufQ);
  k_tconv_glu<64, false><<<nTilesM, 256, 0, stream>>>(bufQ,  wpk22, b2_t2f_b, b2_t2g_b, bufS);
  k_ln<<<nLN, 256, 0, stream>>>(bufS, nullptr, nullptr, nullptr, h1f32, b2_ln_g, b2_ln_b, nullptr, h2f16, 1);

  k_out_gemm<<<(B_ * N_) / 16, 288, 0, stream>>>(h2f16, ow16, out_b, (float*)d_out);
}
